// CONCH_ZeroShot_Model_TopjPooling_MoreText_23304492548567
// MI455X (gfx1250) — matrix operand order repaired and checked by reading, not compiled, not run on hardware
//
#include <hip/hip_runtime.h>
#include <hip/hip_bf16.h>
#include <math.h>

typedef __attribute__((ext_vector_type(16))) __bf16 v16bf;
typedef __attribute__((ext_vector_type(8)))  float  v8f;

#define TCLS  4    // NUM_CLASSES
#define DPC   16   // DESC_PER_CLASS
#define TDESC 64   // TCLS * DPC

// ---------------------------------------------------------------------------
// Kernel A: fused (implicit-normalize) patch x descriptor similarity + scores
//   score[b][p] = max(0, max_t  (x[p] . desc[t])) / ||x[p]||
// bf16 WMMA 16x16x32 with f32 accumulate; ||x|| computed in fp32 on the fly.
// ---------------------------------------------------------------------------
__global__ __launch_bounds__(256)
void conch_score_kernel(const float* __restrict__ xs,
                        const float* __restrict__ xl,
                        const float* __restrict__ desc,
                        float* __restrict__ scores,
                        int N, int D, int rbPerBatch)
{
    // Descriptors in LDS as bf16, row pitch 528 (32B-aligned rows, bank spread)
    __shared__ __bf16 s_desc[TDESC][528];
    const int tid = threadIdx.x;
    for (int i = tid; i < TDESC * D; i += 256) {
        int n = i / D;
        int k = i - n * D;
        s_desc[n][k] = (__bf16)desc[i];
    }
    __syncthreads();

    const int wave = tid >> 5;
    const int lane = tid & 31;
    const int rb   = blockIdx.x * 8 + wave;
    if (rb >= rbPerBatch) return;

    const int b  = blockIdx.y;
    const int p0 = rb * 16;
    const int m  = lane & 15;           // A-matrix row owned by this lane
    const int h  = (lane >> 4) << 3;    // 0 or 8 : K half-offset (A layout)
    const int p  = p0 + m;
    const float* row = (p < N)
        ? xs + ((size_t)b * (size_t)N + (size_t)p)       * (size_t)D
        : xl + ((size_t)b * (size_t)N + (size_t)(p - N)) * (size_t)D;

    v8f acc0 = {}, acc1 = {}, acc2 = {}, acc3 = {};
    float ssq = 0.0f;
    const int kb_lane = (lane >> 4) << 4;   // 0 or 16 : K offset (B layout)
    const int nb      = lane & 15;          // descriptor column owned (B layout)

    for (int k0 = 0; k0 < D; k0 += 32) {
        // A fragment, contiguous K order (matches B):
        //   elems 0..15 -> K = k0 + 2*h + i
        float4 f0 = *(const float4*)(row + k0 + 2 * h);
        float4 f1 = *(const float4*)(row + k0 + 2 * h + 4);
        float4 f2 = *(const float4*)(row + k0 + 2 * h + 8);
        float4 f3 = *(const float4*)(row + k0 + 2 * h + 12);
        ssq += f0.x*f0.x + f0.y*f0.y + f0.z*f0.z + f0.w*f0.w
             + f1.x*f1.x + f1.y*f1.y + f1.z*f1.z + f1.w*f1.w
             + f2.x*f2.x + f2.y*f2.y + f2.z*f2.z + f2.w*f2.w
             + f3.x*f3.x + f3.y*f3.y + f3.z*f3.z + f3.w*f3.w;

        v16bf a;
        a[0]=(__bf16)f0.x;  a[1]=(__bf16)f0.y;  a[2]=(__bf16)f0.z;  a[3]=(__bf16)f0.w;
        a[4]=(__bf16)f1.x;  a[5]=(__bf16)f1.y;  a[6]=(__bf16)f1.z;  a[7]=(__bf16)f1.w;
        a[8]=(__bf16)f2.x;  a[9]=(__bf16)f2.y;  a[10]=(__bf16)f2.z; a[11]=(__bf16)f2.w;
        a[12]=(__bf16)f3.x; a[13]=(__bf16)f3.y; a[14]=(__bf16)f3.z; a[15]=(__bf16)f3.w;

        // B fragments, 16-bit B 32x16 layout: lane holds desc row (nb+tile),
        // contiguous K = k0 + kb_lane + i  (32B aligned in LDS)
        const int kb = k0 + kb_lane;
        v16bf b0 = *(const v16bf*)&s_desc[nb      ][kb];
        v16bf b1 = *(const v16bf*)&s_desc[nb + 16 ][kb];
        v16bf b2 = *(const v16bf*)&s_desc[nb + 32 ][kb];
        v16bf b3 = *(const v16bf*)&s_desc[nb + 48 ][kb];

        acc0 = __builtin_amdgcn_wmma_f32_16x16x32_bf16(false, a, false, b0, (short)0, acc0, false, false);
        acc1 = __builtin_amdgcn_wmma_f32_16x16x32_bf16(false, a, false, b1, (short)0, acc1, false, false);
        acc2 = __builtin_amdgcn_wmma_f32_16x16x32_bf16(false, a, false, b2, (short)0, acc2, false, false);
        acc3 = __builtin_amdgcn_wmma_f32_16x16x32_bf16(false, a, false, b3, (short)0, acc3, false, false);
    }

    // full sum-of-squares for row m: lanes L and L+16 hold complementary K's
    ssq += __shfl_xor(ssq, 16, 32);
    float rinv = 1.0f / fmaxf(sqrtf(ssq), 1e-12f);

    // C/D layout: lane = N column (per half), VGPR r = M row (r or 8+r)
    float rm[8];
    #pragma unroll
    for (int r = 0; r < 8; ++r)
        rm[r] = fmaxf(fmaxf(acc0[r], acc1[r]), fmaxf(acc2[r], acc3[r]));
    #pragma unroll
    for (int mask = 1; mask <= 8; mask <<= 1) {
        #pragma unroll
        for (int r = 0; r < 8; ++r)
            rm[r] = fmaxf(rm[r], __shfl_xor(rm[r], mask, 32));
    }
    // lanes 0..7 write rows 0..7 (half 0); lanes 24..31 write rows 8..15 (half 1)
    float myrm = rm[0];
    #pragma unroll
    for (int r = 1; r < 8; ++r)
        myrm = ((lane & 7) == r) ? rm[r] : myrm;
    if (lane < 8 || lane >= 24) {
        int mrow = (lane < 8) ? lane : (8 + (lane & 7));  // == lane&15, so rinv matches
        scores[(size_t)b * (size_t)(2 * N) + (size_t)(p0 + mrow)] =
            fmaxf(myrm, 0.0f) * rinv;
    }
}

// ---------------------------------------------------------------------------
// Kernel B: per-batch top-k, fp32 pooled feature, class logits, softmax/argmax
// ---------------------------------------------------------------------------
__global__ __launch_bounds__(256)
void conch_finalize_kernel(const float* __restrict__ xs,
                           const float* __restrict__ xl,
                           const float* __restrict__ desc,
                           const float* __restrict__ logit_scale,
                           const int* __restrict__ topj_p,
                           float* __restrict__ scores,
                           float* __restrict__ out,
                           int B, int N, int D)
{
    const int b   = blockIdx.x;
    const int tid = threadIdx.x;
    const int P   = 2 * N;
    int topj = *topj_p;
    if (topj > P)    topj = P;
    if (topj > 1024) topj = 1024;

    __shared__ float s_v[256];
    __shared__ int   s_i[256];
    __shared__ int   s_idx[1024];
    __shared__ float s_tmp[256];
    __shared__ float s_logit[TCLS];

    float* sc = scores + (size_t)b * (size_t)P;

    // ---- iterative top-k extraction (ties -> smallest index, like lax.top_k)
    for (int j = 0; j < topj; ++j) {
        float bv = -INFINITY; int bi = 0x7fffffff;
        for (int pp = tid; pp < P; pp += 256) {
            float v = sc[pp];
            if (v > bv) { bv = v; bi = pp; }
        }
        s_v[tid] = bv; s_i[tid] = bi;
        __syncthreads();
        for (int s = 128; s > 0; s >>= 1) {
            if (tid < s) {
                float ov = s_v[tid + s]; int oi = s_i[tid + s];
                if (ov > s_v[tid] || (ov == s_v[tid] && oi < s_i[tid])) {
                    s_v[tid] = ov; s_i[tid] = oi;
                }
            }
            __syncthreads();
        }
        if (tid == 0) { s_idx[j] = s_i[0]; sc[s_i[0]] = -INFINITY; }
        __syncthreads();
    }

    // ---- sum of L2-normalized selected rows (fp32 exact)
    float accv[4] = {0.f, 0.f, 0.f, 0.f};  // dims tid + q*256 (D <= 1024)
    for (int j = 0; j < topj; ++j) {
        int pp = s_idx[j];
        const float* row = (pp < N)
            ? xs + ((size_t)b * (size_t)N + (size_t)pp)       * (size_t)D
            : xl + ((size_t)b * (size_t)N + (size_t)(pp - N)) * (size_t)D;
        float part = 0.f;
        for (int d = tid; d < D; d += 256) { float v = row[d]; part += v * v; }
        s_tmp[tid] = part;
        __syncthreads();
        for (int s = 128; s > 0; s >>= 1) { if (tid < s) s_tmp[tid] += s_tmp[tid + s]; __syncthreads(); }
        float rinv = 1.0f / fmaxf(sqrtf(s_tmp[0]), 1e-12f);
        __syncthreads();
        #pragma unroll
        for (int q = 0; q < 4; ++q) { int d = tid + q * 256; if (d < D) accv[q] += row[d] * rinv; }
    }

    // ---- image_features = l2norm(sum)  (mean factor cancels)
    float part = 0.f;
    #pragma unroll
    for (int q = 0; q < 4; ++q) part += accv[q] * accv[q];
    s_tmp[tid] = part;
    __syncthreads();
    for (int s = 128; s > 0; s >>= 1) { if (tid < s) s_tmp[tid] += s_tmp[tid + s]; __syncthreads(); }
    float rinv2 = 1.0f / fmaxf(sqrtf(s_tmp[0]), 1e-12f);
    __syncthreads();

    // ---- class logits = img . mean(desc per class)
    float cl[TCLS] = {0.f, 0.f, 0.f, 0.f};
    #pragma unroll
    for (int q = 0; q < 4; ++q) {
        int d = tid + q * 256;
        if (d < D) {
            float img = accv[q] * rinv2;
            #pragma unroll
            for (int c = 0; c < TCLS; ++c) {
                float ct = 0.f;
                for (int t = 0; t < DPC; ++t)
                    ct += desc[(size_t)(c * DPC + t) * (size_t)D + (size_t)d];
                cl[c] += img * (ct * (1.0f / (float)DPC));
            }
        }
    }
    for (int c = 0; c < TCLS; ++c) {
        s_tmp[tid] = cl[c];
        __syncthreads();
        for (int s = 128; s > 0; s >>= 1) { if (tid < s) s_tmp[tid] += s_tmp[tid + s]; __syncthreads(); }
        if (tid == 0) s_logit[c] = s_tmp[0];
        __syncthreads();
    }

    if (tid == 0) {
        float scale = expf(logit_scale[0]);
        float lg[TCLS], mx = -INFINITY;
        for (int c = 0; c < TCLS; ++c) { lg[c] = s_logit[c] * scale; mx = fmaxf(mx, lg[c]); }
        float se = 0.f, e[TCLS];
        for (int c = 0; c < TCLS; ++c) { e[c] = expf(lg[c] - mx); se += e[c]; }
        for (int c = 0; c < TCLS; ++c) out[b * TCLS + c] = e[c] / se;  // Y_prob
        int am = 0; float bvv = lg[0];
        for (int c = 1; c < TCLS; ++c) if (lg[c] > bvv) { bvv = lg[c]; am = c; }
        out[B * TCLS + b] = (float)am;                                  // Y_hat
    }
}

// ---------------------------------------------------------------------------
extern "C" void kernel_launch(void* const* d_in, const int* in_sizes, int n_in,
                              void* d_out, int out_size, void* d_ws, size_t ws_size,
                              hipStream_t stream) {
    (void)n_in; (void)ws_size;
    const float* xs   = (const float*)d_in[0];   // x_s   [B,N,D]
    const float* xl   = (const float*)d_in[2];   // x_l   [B,N,D]
    const float* desc = (const float*)d_in[4];   // desc  [64,D]
    const float* lsc  = (const float*)d_in[5];   // logit_scale [1]
    const int*   topj = (const int*)d_in[6];     // topj scalar
    float* out = (float*)d_out;

    int B = out_size / (TCLS + 1);               // out = B*4 probs + B hats
    long long BN = (long long)in_sizes[1] / 2;   // coord_s = [B,N,2] -> B*N
    int N = (int)(BN / (long long)B);
    int D = (int)((long long)in_sizes[0] / BN);

    float* scores = (float*)d_ws;                // B * 2N floats
    int rbPerBatch = (2 * N) / 16;               // 16-row WMMA blocks per batch
    dim3 gA((rbPerBatch + 7) / 8, B);
    conch_score_kernel<<<gA, 256, 0, stream>>>(xs, xl, desc, scores, N, D, rbPerBatch);
    conch_finalize_kernel<<<dim3(B), 256, 0, stream>>>(xs, xl, desc, lsc, topj,
                                                       scores, out, B, N, D);
}
